// GatherModel_463856468342
// MI455X (gfx1250) — matrix operand order repaired and checked by reading, not compiled, not run on hardware
//
#include <hip/hip_runtime.h>
#include <stdint.h>

// ---------------------------------------------------------------------------
// Problem constants (match reference): N=8000, E=15000, D=128, DE=16, DH=128
// ---------------------------------------------------------------------------
#define D_DIM   128
#define DE_DIM  16
#define DD      16384       // D*D
#define NSTEPS  3

typedef __bf16    bf16x16 __attribute__((ext_vector_type(16)));
typedef float     f32x8   __attribute__((ext_vector_type(8)));
typedef uint32_t  u32x4   __attribute__((ext_vector_type(4)));

union Frag16 { bf16x16 v; uint32_t u[8]; };

__device__ __forceinline__ uint16_t f2bf(float f) {
    uint32_t u = __float_as_uint(f);
    u += 0x7FFFu + ((u >> 16) & 1u);      // round-to-nearest-even
    return (uint16_t)(u >> 16);
}
__device__ __forceinline__ float bf2f(uint16_t h) {
    return __uint_as_float(((uint32_t)h) << 16);
}

// ---------------------------------------------------------------------------
// One-time: repack W2 (fp32 [128,16384]) into this GEMM's B-fragment order (contiguous K per lane half),
// bf16-pair dwords:  w2p[((kbi*1024 + nt)*32 + lane)*8 + p]
//   kbi  = k-block (K=32 each, 4 total)
//   nt   = 16-column tile (1024 total)
//   lane = WMMA lane (l<16: N=l, K=0..15 of block; l>=16: same N, K=16..31)
//   p    = dword pair index, K = {2p, 2p+1} + 16*(lane>>4)
// GEMM then loads each B fragment as 32 contiguous bytes per lane (2x b128).
// ---------------------------------------------------------------------------
__global__ __launch_bounds__(256) void w2_repack_kernel(
    const float* __restrict__ W2, uint32_t* __restrict__ w2p)
{
    int i = blockIdx.x * 256 + threadIdx.x;   // 0 .. 4*1024*32*8-1
    int p    = i & 7;
    int l    = (i >> 3) & 31;
    int nt   = (i >> 8) & 1023;
    int kbi  = i >> 18;
    int half = l >> 4, l16 = l & 15;
    int k0 = kbi * 32 + 2 * p + 16 * half;
    int n  = nt * 16 + l16;
    uint32_t lo = f2bf(W2[(size_t)k0 * DD + n]);
    uint32_t hi = f2bf(W2[(size_t)(k0 + 1) * DD + n]);
    w2p[i] = lo | (hi << 16);
}

// ---------------------------------------------------------------------------
// h = relu(edge_attr @ W1 + b1)  -> bf16 [E, 128]
// ---------------------------------------------------------------------------
__global__ __launch_bounds__(128) void edge_hidden_kernel(
    const float* __restrict__ edge_attr,   // [E,16]
    const float* __restrict__ W1,          // [16,128]
    const float* __restrict__ b1,          // [128]
    uint16_t* __restrict__ h,              // [E,128] bf16
    int E)
{
    int e = blockIdx.x;
    int j = threadIdx.x;
    __shared__ float ea[DE_DIM];
    if (j < DE_DIM) ea[j] = edge_attr[(size_t)e * DE_DIM + j];
    __syncthreads();
    float acc = b1[j];
    #pragma unroll
    for (int k = 0; k < DE_DIM; ++k) acc += ea[k] * W1[k * D_DIM + j];
    h[(size_t)e * D_DIM + j] = f2bf(acc > 0.f ? acc : 0.f);
}

// ---------------------------------------------------------------------------
// Flagship WMMA GEMM:  ew[E,16384] = h[E,128] @ W2[128,16384] + b2  (bf16 out)
// Block = 4 waves (128 threads). Each wave owns a 16(M) x 64(N) output tile
// (4 accumulators), K = 128 in four 16x16x32 bf16 WMMA steps.
// A: 2x b128 per fragment (K-pairs contiguous in h).
// B: 2x b128 per fragment from pre-repacked w2p.
// C: bf16 through LDS, stored as row-contiguous b128 (coalesced 128B rows).
// ---------------------------------------------------------------------------
__global__ __launch_bounds__(128) void ew_gemm_kernel(
    const uint16_t* __restrict__ h,    // [E,128] bf16 row-major
    const uint32_t* __restrict__ w2p,  // repacked B fragments
    const float*    __restrict__ b2,   // [16384]
    uint16_t*       __restrict__ ew,   // [E,16384] bf16 row-major
    int E)
{
    const int tid  = threadIdx.x;
    const int lane = tid & 31;
    const int wave = tid >> 5;
    const int half = lane >> 4;     // 0: lanes 0-15, 1: lanes 16-31
    const int l16  = lane & 15;

    const int mBase = blockIdx.x * 16;
    const int nWave = blockIdx.y * 256 + wave * 64;

    const int row  = mBase + l16;
    const int rowc = row < E ? row : (E - 1);   // clamp A loads for M remainder

    f32x8 acc[4] = {};

    #pragma unroll
    for (int kbi = 0; kbi < 4; ++kbi) {
        const int kb = kbi * 32;
        // ---- A fragment: 16x32 bf16 (lanes=M; half selects K+16)
        // pair p -> K = {2p,2p+1}+16*half
        Frag16 a;
        const uint16_t* hrow = h + (size_t)rowc * D_DIM + kb + 16 * half;
        #pragma unroll
        for (int p = 0; p < 8; ++p) {
            int koff = 2 * p;
            a.u[p] = *(const uint32_t*)(hrow + koff);   // -> 2x global_load_b128
        }
        #pragma unroll
        for (int t = 0; t < 4; ++t) {
            const int nt = blockIdx.y * 16 + wave * 4 + t;
            const uint32_t* bp = w2p + (((size_t)kbi * 1024 + nt) * 32 + lane) * 8;
            Frag16 b;
            #pragma unroll
            for (int p = 0; p < 8; ++p) b.u[p] = bp[p];  // -> 2x global_load_b128
            acc[t] = __builtin_amdgcn_wmma_f32_16x16x32_bf16(
                false, a.v, false, b.v, (short)0, acc[t], false, false);
        }
    }

    // ---- epilogue: +b2, fp32->bf16, repack via LDS, coalesced b128 row stores
    __shared__ __align__(16) uint16_t st[4 * 16 * 64];   // per-wave 16x64 bf16
    #pragma unroll
    for (int t = 0; t < 4; ++t) {
        const int n = nWave + t * 16 + l16;
        const float bb = b2[n];
        #pragma unroll
        for (int r = 0; r < 8; ++r)
            st[wave * 1024 + (r + 8 * half) * 64 + t * 16 + l16] =
                f2bf(acc[t][r] + bb);
    }
    __syncthreads();
    const u32x4* lsrc = (const u32x4*)(st + wave * 1024);  // 128 chunks of 16B
    #pragma unroll
    for (int it = 0; it < 4; ++it) {
        const int idx   = lane + 32 * it;      // 0..127
        const int rowi  = idx >> 3;            // 0..15
        const int chunk = idx & 7;             // 8 x 16B per 128B row segment
        const int rr = mBase + rowi;
        if (rr < E)
            *(u32x4*)(ew + (size_t)rr * DD + nWave + chunk * 8) =
                lsrc[rowi * 8 + chunk];
    }
}

// ---------------------------------------------------------------------------
// out0 = relu(x @ W0 + b0)   — one 128-thread block per node row
// ---------------------------------------------------------------------------
__global__ __launch_bounds__(128) void init_kernel(
    const float* __restrict__ x, const float* __restrict__ W0,
    const float* __restrict__ b0, float* __restrict__ out, int N)
{
    int n = blockIdx.x, j = threadIdx.x;
    __shared__ float s[D_DIM];
    s[j] = x[(size_t)n * D_DIM + j];
    __syncthreads();
    float acc = b0[j];
    #pragma unroll 8
    for (int k = 0; k < D_DIM; ++k) acc += s[k] * W0[k * D_DIM + j];
    out[(size_t)n * D_DIM + j] = acc > 0.f ? acc : 0.f;
}

// ---------------------------------------------------------------------------
// Per-edge matvec + scatter-add:  agg[dst] += out[src] @ ew[e]
// 64-thread block per edge; each thread owns 2 outputs (one dword of bf16).
// Streams the 32KB ew row fully coalesced (256B per d-iteration).
// ---------------------------------------------------------------------------
__global__ __launch_bounds__(64) void edge_msg_kernel(
    const float* __restrict__ outv, const uint16_t* __restrict__ ew,
    const int* __restrict__ ei, float* __restrict__ agg, int E)
{
    int e = blockIdx.x, t = threadIdx.x;
    int src = ei[e];
    int dst = ei[E + e];
    __shared__ float s[D_DIM];
    s[t]      = outv[(size_t)src * D_DIM + t];
    s[t + 64] = outv[(size_t)src * D_DIM + t + 64];
    __syncthreads();
    const uint32_t* w = (const uint32_t*)(ew + (size_t)e * DD);
    float a0 = 0.f, a1 = 0.f;
    #pragma unroll 4
    for (int d = 0; d < D_DIM; ++d) {
        uint32_t pk = w[d * 64 + t];           // elems (2t, 2t+1) of row d
        float sd = s[d];
        a0 += sd * bf2f((uint16_t)pk);
        a1 += sd * bf2f((uint16_t)(pk >> 16));
    }
    atomicAdd(&agg[(size_t)dst * D_DIM + 2 * t    ], a0);
    atomicAdd(&agg[(size_t)dst * D_DIM + 2 * t + 1], a1);
}

// ---------------------------------------------------------------------------
// Fused per-step node update:
//   m   = relu(agg + out @ root + bias)
//   out'= m @ Wm[0:128] + out @ Wm[128:256] + bm   (+ x on last step)
// ---------------------------------------------------------------------------
__global__ __launch_bounds__(128) void combine_kernel(
    const float* __restrict__ cur,  const float* __restrict__ agg,
    const float* __restrict__ root, const float* __restrict__ bias,
    const float* __restrict__ Wm,   const float* __restrict__ bm,
    const float* __restrict__ x,    float* __restrict__ nxt,
    int N, int addInit)
{
    int n = blockIdx.x, j = threadIdx.x;
    __shared__ float so[D_DIM];
    __shared__ float sm[D_DIM];
    so[j] = cur[(size_t)n * D_DIM + j];
    __syncthreads();

    float a = agg[(size_t)n * D_DIM + j] + bias[j];
    #pragma unroll 8
    for (int k = 0; k < D_DIM; ++k) a += so[k] * root[k * D_DIM + j];
    sm[j] = a > 0.f ? a : 0.f;
    __syncthreads();

    float o = bm[j];
    #pragma unroll 4
    for (int k = 0; k < D_DIM; ++k) {
        o += sm[k] * Wm[k * D_DIM + j];
        o += so[k] * Wm[(D_DIM + k) * D_DIM + j];
    }
    if (addInit) o += x[(size_t)n * D_DIM + j];
    nxt[(size_t)n * D_DIM + j] = o;
}

// ---------------------------------------------------------------------------
extern "C" void kernel_launch(void* const* d_in, const int* in_sizes, int n_in,
                              void* d_out, int out_size, void* d_ws, size_t ws_size,
                              hipStream_t stream)
{
    const float* x    = (const float*)d_in[0];
    const float* eatt = (const float*)d_in[1];
    const float* W0   = (const float*)d_in[2];
    const float* b0   = (const float*)d_in[3];
    const float* W1   = (const float*)d_in[4];
    const float* b1   = (const float*)d_in[5];
    const float* W2   = (const float*)d_in[6];
    const float* b2   = (const float*)d_in[7];
    const float* root = (const float*)d_in[8];
    const float* bias = (const float*)d_in[9];
    const float* Wm   = (const float*)d_in[10];
    const float* bm   = (const float*)d_in[11];
    const int*   eidx = (const int*)d_in[12];

    const int N = in_sizes[0] / D_DIM;   // 8000
    const int E = in_sizes[1] / DE_DIM;  // 15000

    // workspace carve-up (256B aligned)
    char* ws = (char*)d_ws;
    size_t off = 0;
    auto carve = [&](size_t bytes) {
        void* p = ws + off;
        off += (bytes + 255) & ~(size_t)255;
        return p;
    };
    uint16_t* ew_bf = (uint16_t*)carve((size_t)E * DD * sizeof(uint16_t));     // 492 MB
    uint16_t* h_bf  = (uint16_t*)carve((size_t)E * D_DIM * sizeof(uint16_t));
    uint32_t* w2p   = (uint32_t*)carve((size_t)(D_DIM / 2) * DD * sizeof(uint32_t));
    float*    outA  = (float*)carve((size_t)N * D_DIM * sizeof(float));
    float*    outB  = (float*)carve((size_t)N * D_DIM * sizeof(float));
    float*    agg   = (float*)carve((size_t)N * D_DIM * sizeof(float));

    // one-time: B repack (incl. bf16 convert), hidden layer, big WMMA GEMM
    w2_repack_kernel<<<(4 * 1024 * 32 * 8) / 256, 256, 0, stream>>>(W2, w2p);
    edge_hidden_kernel<<<E, 128, 0, stream>>>(eatt, W1, b1, h_bf, E);
    ew_gemm_kernel<<<dim3((E + 15) / 16, DD / 256), 128, 0, stream>>>(h_bf, w2p, b2, ew_bf, E);

    // out0 = relu(x @ W0 + b0)
    init_kernel<<<N, 128, 0, stream>>>(x, W0, b0, outA, N);

    float* cur = outA;
    for (int step = 0; step < NSTEPS; ++step) {
        hipMemsetAsync(agg, 0, (size_t)N * D_DIM * sizeof(float), stream);
        edge_msg_kernel<<<E, 64, 0, stream>>>(cur, ew_bf, eidx, agg, E);
        const int last = (step == NSTEPS - 1);
        float* nxt = last ? (float*)d_out : (cur == outA ? outB : outA);
        combine_kernel<<<N, 128, 0, stream>>>(cur, agg, root, bias, Wm, bm, x, nxt, N, last);
        cur = nxt;
    }
}
